// DecoderModel_48954037240034
// MI455X (gfx1250) — matrix operand order repaired and checked by reading, not compiled, not run on hardware
//
#include <hip/hip_runtime.h>
#include <math.h>

// ---------------------------------------------------------------------------
// DCRNN decoder step for MI455X (gfx1250, wave32, WMMA + TDM).
// Heavy ops: 16x  (4096x4096) x (4096 x B*Dp) GEMMs  -> v_wmma_f32_16x16x32_bf16
//            20x  (131072 x Dp) x (Dp x 128) projection GEMMs (accumulating)
// S matrices converted once to bf16 (67 MB total -> resident in 192 MB L2).
// A-tiles staged into LDS by the Tensor Data Mover (tensor_load_to_lds),
// double-buffered, synchronized via s_wait_tensorcnt.
// ---------------------------------------------------------------------------

#define Nn  4096
#define Bb  32
#define Uu  64
#define NBr (Nn * Bb)          // 131072 (node-major rows: n*B + b)

typedef __attribute__((ext_vector_type(16))) __bf16 v16bf;
typedef __attribute__((ext_vector_type(8)))  float  v8f;
typedef __attribute__((ext_vector_type(4)))  unsigned int v4u;
typedef __attribute__((ext_vector_type(8)))  int v8i;
typedef __attribute__((ext_vector_type(4)))  int v4i;

union FragB { uint4 q[2]; v16bf v; };

__device__ __forceinline__ unsigned short f2bf(float f) {
  union { float f; unsigned int u; } x; x.f = f;
  unsigned int r = x.u + 0x7FFFu + ((x.u >> 16) & 1u);   // round-nearest-even
  return (unsigned short)(r >> 16);
}
__device__ __forceinline__ float bf2f(unsigned short h) {
  union { unsigned int u; float f; } x; x.u = ((unsigned int)h) << 16;
  return x.f;
}

#define BM 128
#define BN 128
#define BK 32
#define LDSK 40   // padded K-stride in ushorts; 80B row stride (16B aligned)

// ---------------------------------------------------------------------------
// TDM: load one 128x32 bf16 tile (row stride = ldaElems) into LDS with
// hardware padding of 4 DWORDs (8 ushorts) after every 16 DWORDs (one 64B
// row) -> LDS row stride = 40 ushorts = LDSK.  Descriptor per ISA 8.3/8.4:
//   group0: count=1 | lds_addr | global_addr | type=2
//   group1: data_size=1(2B), pad_enable, pad_interval=3(16 DW),
//           pad_amount=3(4 DW), tensor dims huge (no OOB), tile 32x128,
//           dim0 stride = ldaElems.
// This toolchain exposes the 6-arg builtin (extra int32x8 group, zero-filled).
// ---------------------------------------------------------------------------
__device__ __forceinline__ void tdm_load_tile_a(const unsigned short* gsrc,
                                                unsigned ldsAddr, long ldaElems) {
  unsigned long long ga = (unsigned long long)(uintptr_t)gsrc;
  unsigned long long st = (unsigned long long)ldaElems;
  const unsigned td0 = 0x7FFFFFFFu, td1 = 0x7FFFFFFFu;
  v4u g0;
  g0[0] = 1u;                                             // count = 1
  g0[1] = ldsAddr;                                        // LDS byte address
  g0[2] = (unsigned)(ga & 0xFFFFFFFFull);
  g0[3] = (unsigned)((ga >> 32) & 0x01FFFFFFull) | (2u << 30);  // type = 2
  v8i g1;
  g1[0] = (int)((1u << 16) | (1u << 20) | (3u << 22) | (3u << 25));
  g1[1] = (int)((td0 & 0xFFFFu) << 16);                   // tensor_dim0 lo
  g1[2] = (int)(((td0 >> 16) & 0xFFFFu) | ((td1 & 0xFFFFu) << 16));
  g1[3] = (int)(((td1 >> 16) & 0xFFFFu) | (32u << 16));   // tile_dim0 = 32
  g1[4] = 128;                                            // tile_dim1 = 128
  g1[5] = (int)(st & 0xFFFFFFFFull);                      // dim0 stride lo
  g1[6] = (int)((st >> 32) & 0xFFFFull);                  // stride hi, dim1 stride=0
  g1[7] = 0;
  v4i g2 = {0, 0, 0, 0};
  v4i g3 = {0, 0, 0, 0};
  v8i g4 = {0, 0, 0, 0, 0, 0, 0, 0};
  __builtin_amdgcn_tensor_load_to_lds(g0, g1, g2, g3, g4, 0);
}

// ---------------------------------------------------------------------------
// Generic bf16 WMMA GEMM:  out = alpha * A(MxK) @ B(KxN)  [- Sub]  [+ Cf_old]
// Tile: 128x128, BK=32, 256 threads = 8 waves (2x4), each wave 64x32 = 4x2
// frags.  Requires K % 32 == 0, Ncol % 8 == 0, M % 128 == 0 (TDM tile load
// is unguarded) — true for all call sites here.
// ---------------------------------------------------------------------------
__global__ __launch_bounds__(256)
void wmma_gemm_bf16(const unsigned short* __restrict__ A, long lda,
                    const unsigned short* __restrict__ Bm, long ldb,
                    float alpha,
                    const unsigned short* __restrict__ Sub, long ldsb,
                    float* __restrict__ Cf, long ldc, int accum,
                    unsigned short* __restrict__ Cb, long ldcb,
                    int M, int K, int Ncol)
{
  __shared__ alignas(16) unsigned short sA[2][BM * LDSK];
  __shared__ alignas(16) unsigned short sB[2][BN * LDSK];  // transposed [n][k]

  const int tid  = threadIdx.x;
  const int lane = tid & 31;
  const int wv   = tid >> 5;
  const int wr   = wv >> 2;      // wave row 0..1 (64 M-rows each)
  const int wc   = wv & 3;       // wave col 0..3 (32 N-cols each)
  const int lh   = lane >> 4;    // lane half (selects K sub-range per ISA layout)
  const int ll   = lane & 15;
  const long blockM = (long)blockIdx.y * BM;
  const long blockN = (long)blockIdx.x * BN;

  v8f acc[4][2];
#pragma unroll
  for (int i = 0; i < 4; ++i)
#pragma unroll
    for (int j = 0; j < 2; ++j)
      acc[i][j] = (v8f){0.f,0.f,0.f,0.f,0.f,0.f,0.f,0.f};

  const int ntk = K / BK;
  // prologue: TDM fetch of A tile 0 (wave 0 issues; TDM is per-wave, EXEC-free)
  if (wv == 0)
    tdm_load_tile_a(A + blockM * lda, (unsigned)(uintptr_t)&sA[0][0], lda);

  for (int t = 0; t < ntk; ++t) {
    const int k0  = t * BK;
    const int buf = t & 1;

    // ---- stage B tile (32 x 128) into LDS transposed [n][k] ----
#pragma unroll
    for (int i = 0; i < 2; ++i) {
      int f  = tid + 256 * i;
      int kr = f >> 4;                // 16 uint4 per B row of 128
      int c8 = (f & 15) << 3;
      long gc = blockN + c8;
      uint4 v = make_uint4(0, 0, 0, 0);
      if (gc < Ncol) {
        v = *(const uint4*)(Bm + (long)(k0 + kr) * ldb + gc);
        if (k0 + BK < K)  // -> global_prefetch_b8
          __builtin_prefetch(Bm + (long)(k0 + BK + kr) * ldb + gc, 0, 1);
      }
      const unsigned short* pv = (const unsigned short*)&v;
#pragma unroll
      for (int j = 0; j < 8; ++j) sB[buf][(c8 + j) * LDSK + kr] = pv[j];
    }

    // ---- issue TDM for next A tile into the other buffer ----
    if (wv == 0 && t + 1 < ntk)
      tdm_load_tile_a(A + blockM * lda + (k0 + BK),
                      (unsigned)(uintptr_t)&sA[(t + 1) & 1][0], lda);

    // wave0: ensure A tile t has landed (<=1 outstanding keeps t+1 in flight)
    if (t + 1 < ntk) __builtin_amdgcn_s_wait_tensorcnt(1);
    else             __builtin_amdgcn_s_wait_tensorcnt(0);
    __syncthreads();

    // ---- WMMA: one 16x16x32 per (fm,fn) ----
#pragma unroll
    for (int fm = 0; fm < 4; ++fm) {
      // A frag (ISA 16-bit 16x32): lanes 0-15 row M=ll get K0-7 & K16-23,
      // lanes 16-31 get K8-15 & K24-31.
      int ar = wr * 64 + fm * 16 + ll;
      FragB fa;
      fa.q[0] = *(const uint4*)(sA[buf] + ar * LDSK + lh * 8);
      fa.q[1] = *(const uint4*)(sA[buf] + ar * LDSK + lh * 8 + 16);
#pragma unroll
      for (int fn = 0; fn < 2; ++fn) {
        // B frag: lanes 0-15 col N=ll get K0-7 & K16-23; lanes 16-31 get K8-15 & K24-31.
        int bc = wc * 32 + fn * 16 + ll;
        FragB fb;
        fb.q[0] = *(const uint4*)(sB[buf] + bc * LDSK + lh * 8);
        fb.q[1] = *(const uint4*)(sB[buf] + bc * LDSK + lh * 8 + 16);
        acc[fm][fn] = __builtin_amdgcn_wmma_f32_16x16x32_bf16(
            false, fa.v, false, fb.v, (short)0, acc[fm][fn], false, false);
      }
    }
    __syncthreads();   // buffers free for the next TDM / B stores
  }

  // ---- epilogue: C/D layout = VGPR r -> row r (lanes<16) / r+8, col = ll ----
#pragma unroll
  for (int fm = 0; fm < 4; ++fm) {
    long grow0 = blockM + wr * 64 + fm * 16 + lh * 8;
#pragma unroll
    for (int fn = 0; fn < 2; ++fn) {
      long gcol = blockN + wc * 32 + fn * 16 + ll;
      if (gcol >= Ncol) continue;
#pragma unroll
      for (int r = 0; r < 8; ++r) {
        long grow = grow0 + r;
        if (grow >= M) continue;
        float v = alpha * acc[fm][fn][r];
        if (Sub) v -= bf2f(Sub[grow * ldsb + gcol]);
        if (Cf) {
          if (accum) v += Cf[grow * ldc + gcol];
          Cf[grow * ldc + gcol] = v;
        }
        if (Cb) Cb[grow * ldcb + gcol] = f2bf(v);
      }
    }
  }
}

// ---------------------------------------------------------------------------
// Elementwise helpers
// ---------------------------------------------------------------------------
__global__ void k_f32_to_bf16(const float* __restrict__ src,
                              unsigned short* __restrict__ dst, long n) {
  long i = (long)blockIdx.x * blockDim.x + threadIdx.x;
  if (i < n) dst[i] = f2bf(src[i]);
}

// Pack 5 Chebyshev weight slices W[m*D:(m+1)*D, :Wc] into bf16 buffers of
// fixed shape 128x128 (K padded to Dp with zero rows, cols padded to 128).
__global__ void k_pack_weights(const float* __restrict__ W, int D, int Wc,
                               int Dp, unsigned short* __restrict__ out) {
  long i = (long)blockIdx.x * blockDim.x + threadIdx.x;
  long total = 5L * Dp * 128;
  if (i >= total) return;
  int m = (int)(i / (Dp * 128));
  int r = (int)((i / 128) % Dp);
  int c = (int)(i & 127);
  float v = 0.f;
  if (r < D && c < Wc) v = W[((long)m * D + r) * Wc + c];
  out[(long)m * 128 * 128 + (long)r * 128 + c] = f2bf(v);
}

// Z[(n*B+b)*Dp + j] = concat(x, (r?)*h) padded with zeros to Dp, bf16.
__global__ void k_pack_z0(unsigned short* __restrict__ Z, int Dp, int xw,
                          const float* __restrict__ X, const float* __restrict__ H,
                          const float* __restrict__ R /* gate r, NBr x 128, or null */) {
  long i = (long)blockIdx.x * blockDim.x + threadIdx.x;
  long total = (long)NBr * Dp;
  if (i >= total) return;
  long row = i / Dp; int j = (int)(i % Dp);
  int n = (int)(row / Bb), b = (int)(row % Bb);
  float v = 0.f;
  if (j < xw) {
    v = X[((long)b * Nn + n) * xw + j];
  } else if (j < xw + Uu) {
    int u = j - xw;
    float h = H[((long)b * Nn + n) * Uu + u];
    if (R) h *= R[row * 128 + u];
    v = h;
  }
  Z[i] = f2bf(v);
}

__global__ void k_sigmoid_bias(float* __restrict__ G, const float* __restrict__ bias) {
  long i = (long)blockIdx.x * blockDim.x + threadIdx.x;
  if (i >= (long)NBr * 128) return;
  int c = (int)(i & 127);
  G[i] = 1.0f / (1.0f + __expf(-(G[i] + bias[c])));
}

__global__ void k_gru_update(const float* __restrict__ G, const float* __restrict__ Cc,
                             const float* __restrict__ bc, const float* __restrict__ H,
                             float* __restrict__ Hout) {
  long i = (long)blockIdx.x * blockDim.x + threadIdx.x;
  if (i >= (long)NBr * Uu) return;
  long row = i / Uu; int u = (int)(i % Uu);
  int n = (int)(row / Bb), b = (int)(row % Bb);
  float ug = G[row * 128 + 64 + u];                       // update gate (sigmoid'd)
  float c  = tanhf(Cc[row * 128 + u] + bc[u]);            // candidate
  long hidx = ((long)b * Nn + n) * Uu + u;
  Hout[hidx] = ug * H[hidx] + (1.0f - ug) * c;
}

__global__ void k_predict(const float* __restrict__ H1, const float* __restrict__ Wp,
                          const float* __restrict__ bp, float* __restrict__ out) {
  long i = (long)blockIdx.x * blockDim.x + threadIdx.x;
  if (i >= (long)Bb * Nn) return;
  float s = bp[0];
  const float* h = H1 + i * Uu;
#pragma unroll
  for (int u = 0; u < Uu; ++u) s += h[u] * Wp[u];
  out[i] = s;
}

// ---------------------------------------------------------------------------
// Host orchestration
// ---------------------------------------------------------------------------
static inline void launch_gemm(hipStream_t s,
                               const unsigned short* A, long lda,
                               const unsigned short* B, long ldb, float alpha,
                               const unsigned short* Sub, long ldsb,
                               float* Cf, long ldc, int accum,
                               unsigned short* Cb, long ldcb,
                               int M, int K, int Ncol) {
  dim3 g((Ncol + BN - 1) / BN, (M + BM - 1) / BM);
  wmma_gemm_bf16<<<g, 256, 0, s>>>(A, lda, B, ldb, alpha, Sub, ldsb,
                                   Cf, ldc, accum, Cb, ldcb, M, K, Ncol);
}

extern "C" void kernel_launch(void* const* d_in, const int* in_sizes, int n_in,
                              void* d_out, int out_size, void* d_ws, size_t ws_size,
                              hipStream_t stream) {
  const float* inputs = (const float*)d_in[0];   // (B,N,1)
  const float* hidden = (const float*)d_in[1];   // (L,B,N,U)
  const float* Sf     = (const float*)d_in[2];   // (N,N)
  const float* Sb     = (const float*)d_in[3];   // (N,N)
  const float* Wg[2]  = {(const float*)d_in[4], (const float*)d_in[8]};
  const float* bg[2]  = {(const float*)d_in[5], (const float*)d_in[9]};
  const float* Wc[2]  = {(const float*)d_in[6], (const float*)d_in[10]};
  const float* bc[2]  = {(const float*)d_in[7], (const float*)d_in[11]};
  const float* Wp     = (const float*)d_in[12];
  const float* bp     = (const float*)d_in[13];

  float* predict = (float*)d_out;                // B*N
  float* hid_out = predict + (long)Bb * Nn;      // (L,B,N,U)

  // workspace carve (256B aligned)
  uintptr_t w = (uintptr_t)d_ws;
  auto carve = [&](size_t bytes) -> void* {
    void* p = (void*)w;
    w += (bytes + 255) & ~(size_t)255;
    return p;
  };
  unsigned short* Sfb  = (unsigned short*)carve((size_t)Nn * Nn * 2);
  unsigned short* Sbb  = (unsigned short*)carve((size_t)Nn * Nn * 2);
  unsigned short* Wgpb = (unsigned short*)carve((size_t)5 * 128 * 128 * 2);
  unsigned short* Wcpb = (unsigned short*)carve((size_t)5 * 128 * 128 * 2);
  unsigned short* xs0  = (unsigned short*)carve((size_t)NBr * 128 * 2);
  unsigned short* xs1  = (unsigned short*)carve((size_t)NBr * 128 * 2);
  unsigned short* xs2  = (unsigned short*)carve((size_t)NBr * 128 * 2);
  float* G  = (float*)carve((size_t)NBr * 128 * 4);
  float* Cc = (float*)carve((size_t)NBr * 128 * 4);

  auto g1 = [](long n) { return dim3((unsigned)((n + 255) / 256)); };

  // Convert supports to bf16 once per call (they then live in L2: 67 MB total).
  k_f32_to_bf16<<<g1((long)Nn * Nn), 256, 0, stream>>>(Sf, Sfb, (long)Nn * Nn);
  k_f32_to_bf16<<<g1((long)Nn * Nn), 256, 0, stream>>>(Sb, Sbb, (long)Nn * Nn);

  for (int l = 0; l < 2; ++l) {
    const int D  = (l == 0) ? (1 + Uu) : (2 * Uu);   // 65 / 128
    const int Dp = (l == 0) ? 96 : 128;              // K padded to mult of 32
    const int xw = (l == 0) ? 1 : Uu;
    const int C  = Bb * Dp;                          // 3072 / 4096 columns
    const float* X = (l == 0) ? inputs : hid_out;    // layer1 input = h0_new
    const float* H = hidden + (size_t)l * Bb * Nn * Uu;

    k_pack_weights<<<g1(5L * Dp * 128), 256, 0, stream>>>(Wg[l], D, 2 * Uu, Dp, Wgpb);
    k_pack_weights<<<g1(5L * Dp * 128), 256, 0, stream>>>(Wc[l], D, Uu, Dp, Wcpb);

    for (int pass = 0; pass < 2; ++pass) {           // 0 = gates, 1 = candidate
      const float* Rg = (pass == 0) ? nullptr : G;   // candidate uses r*h
      unsigned short* Wpk = (pass == 0) ? Wgpb : Wcpb;
      float* Acc = (pass == 0) ? G : Cc;
      const size_t ws_m = (size_t)128 * 128;

      k_pack_z0<<<g1((long)NBr * Dp), 256, 0, stream>>>(xs0, Dp, xw, X, H, Rg);

      // m=0: projection of x0
      launch_gemm(stream, xs0, Dp, Wpk, 128, 1.f, nullptr, 0, Acc, 128, 0,
                  nullptr, 0, NBr, Dp, 128);
      // forward support: x1 = Sf@x0 ; x2 = 2*Sf@x1 - x0
      launch_gemm(stream, Sfb, Nn, xs0, C, 1.f, nullptr, 0, nullptr, 0, 0,
                  xs1, C, Nn, Nn, C);
      launch_gemm(stream, xs1, Dp, Wpk + 1 * ws_m, 128, 1.f, nullptr, 0,
                  Acc, 128, 1, nullptr, 0, NBr, Dp, 128);
      launch_gemm(stream, Sfb, Nn, xs1, C, 2.f, xs0, C, nullptr, 0, 0,
                  xs2, C, Nn, Nn, C);
      launch_gemm(stream, xs2, Dp, Wpk + 2 * ws_m, 128, 1.f, nullptr, 0,
                  Acc, 128, 1, nullptr, 0, NBr, Dp, 128);
      // backward support
      launch_gemm(stream, Sbb, Nn, xs0, C, 1.f, nullptr, 0, nullptr, 0, 0,
                  xs1, C, Nn, Nn, C);
      launch_gemm(stream, xs1, Dp, Wpk + 3 * ws_m, 128, 1.f, nullptr, 0,
                  Acc, 128, 1, nullptr, 0, NBr, Dp, 128);
      launch_gemm(stream, Sbb, Nn, xs1, C, 2.f, xs0, C, nullptr, 0, 0,
                  xs2, C, Nn, Nn, C);
      launch_gemm(stream, xs2, Dp, Wpk + 4 * ws_m, 128, 1.f, nullptr, 0,
                  Acc, 128, 1, nullptr, 0, NBr, Dp, 128);

      if (pass == 0)
        k_sigmoid_bias<<<g1((long)NBr * 128), 256, 0, stream>>>(G, bg[l]);
    }
    k_gru_update<<<g1((long)NBr * Uu), 256, 0, stream>>>(
        G, Cc, bc[l], H, hid_out + (size_t)l * Bb * Nn * Uu);
  }

  k_predict<<<g1((long)Bb * Nn), 256, 0, stream>>>(
      hid_out + (size_t)1 * Bb * Nn * Uu, Wp, bp, predict);
}
